// ResBlock_77129022701583
// MI455X (gfx1250) — matrix operand order repaired and checked by reading, not compiled, not run on hardware
//
#include <hip/hip_runtime.h>
#include <hip/hip_bf16.h>
#include <math.h>

#define B_    4
#define C_    256
#define N_    4096
#define K_    5
#define O_    256
#define TWO_C 512
#define BN_EPS 1e-5f

typedef __attribute__((ext_vector_type(16))) __bf16 v16bf;
typedef __attribute__((ext_vector_type(8)))  float  v8f;

union FragBF { uint4 q[2]; v16bf v; };

__device__ __forceinline__ unsigned short f2bf(float f) {
  union { float f; unsigned u; } x; x.f = f;
  unsigned r = x.u + 0x7FFFu + ((x.u >> 16) & 1u);  // round-to-nearest-even
  return (unsigned short)(r >> 16);
}

// A-matrix 16x32 bf16 fragment (ISA 7.12.2): lanes 0-15 -> K 0..7 & 16..23,
// lanes 16-31 -> K 8..15 & 24..31, rows M = 0..15 on lane&15.
__device__ __forceinline__ FragBF load_a_frag(const unsigned short* base, int row,
                                              int kb, int lane) {
  int half = lane >> 4;
  const unsigned short* p = base + (size_t)row * C_ + kb + half * 8;
  FragBF f;
  f.q[0] = *reinterpret_cast<const uint4*>(p);
  f.q[1] = *reinterpret_cast<const uint4*>(p + 16);
  return f;
}

// B-matrix 32x16 bf16 fragment: lanes 0-15 hold K 0..7 & 16..23, lanes 16-31 K 8..15 & 24..31,
// output col N = lane&15. Element (k, n) = src_rowmajor[n][k].
__device__ __forceinline__ FragBF load_b_frag(const unsigned short* base, int colrow,
                                              int kb, int lane) {
  int half = lane >> 4;
  const unsigned short* p = base + (size_t)colrow * C_ + kb + half * 8;
  FragBF f;
  f.q[0] = *reinterpret_cast<const uint4*>(p);
  f.q[1] = *reinterpret_cast<const uint4*>(p + 16);
  return f;
}

// ---- prep: x (B,C,N) f32 -> xt (B,N,C) bf16 via LDS tile transpose ----
__global__ void k_xt(const float* __restrict__ x, unsigned short* __restrict__ xt) {
  __shared__ float tile[32][33];
  int b = blockIdx.z, n0 = blockIdx.x * 32, c0 = blockIdx.y * 32;
  int tx = threadIdx.x, ty = threadIdx.y;
  for (int j = 0; j < 32; j += 8)
    tile[ty + j][tx] = x[((size_t)b * C_ + c0 + ty + j) * N_ + n0 + tx];
  __syncthreads();
  for (int j = 0; j < 32; j += 8)
    xt[((size_t)b * N_ + n0 + ty + j) * C_ + c0 + tx] = f2bf(tile[tx][ty + j]);
}

// ---- prep: xx[b,n] = sum_c x^2 ----
__global__ void k_xx(const float* __restrict__ x, float* __restrict__ xx) {
  int i = blockIdx.x * 256 + threadIdx.x;   // b*N + n
  int b = i >> 12, n = i & (N_ - 1);
  const float* p = x + (size_t)b * C_ * N_ + n;
  float s = 0.f;
  for (int c = 0; c < C_; ++c) { float v = p[(size_t)c * N_]; s += v * v; }
  xx[i] = s;
}

// ---- prep: V = W1 (bf16), U = W2 - W1 (bf16) ----
__global__ void k_w(const float* __restrict__ W, unsigned short* __restrict__ Vb,
                    unsigned short* __restrict__ Ub) {
  int o = blockIdx.x, c = threadIdx.x;
  float w1 = W[(size_t)o * TWO_C + c];
  float w2 = W[(size_t)o * TWO_C + C_ + c];
  Vb[o * C_ + c] = f2bf(w1);
  Ub[o * C_ + c] = f2bf(w2 - w1);
}

__global__ void k_zero(float* __restrict__ p, int n) {
  int i = blockIdx.x * blockDim.x + threadIdx.x;
  if (i < n) p[i] = 0.f;
}

// ---- fused distance GEMM (bf16 WMMA) + top-5 per row ----
__global__ void __launch_bounds__(512) k_dist_topk(
    const unsigned short* __restrict__ xt, const float* __restrict__ xx,
    int* __restrict__ idxo) {
  __shared__ float strip[16 * 512];               // 32 KB score staging
  __shared__ float candV[16][32][5];              // 10 KB
  __shared__ int   candI[16][32][5];              // 10 KB
  __shared__ unsigned short aRows[16 * C_];       //  8 KB shared A rows
  int b    = blockIdx.x / (N_ / 16);
  int n0   = (blockIdx.x % (N_ / 16)) * 16;
  int wave = threadIdx.x >> 5, lane = threadIdx.x & 31;
  int lr = lane & 15, half = lane >> 4;
  const unsigned short* xtb = xt + (size_t)b * N_ * C_;

  // Async-stage the 16 A rows (8 KB) into LDS once for the whole workgroup:
  // one global_load_async_to_lds_b128 per thread (ASYNCcnt path).
  {
    int t   = threadIdx.x;        // 0..511
    int row = t >> 5, seg = t & 31;
    const unsigned short* gp = xtb + (size_t)(n0 + row) * C_ + seg * 8;
    unsigned lds = (unsigned)(size_t)&aRows[row * C_ + seg * 8];
    unsigned long long ga = (unsigned long long)(size_t)gp;
    asm volatile("global_load_async_to_lds_b128 %0, %1, off"
                 :: "v"(lds), "v"(ga) : "memory");
    asm volatile("s_wait_asynccnt 0x0" ::: "memory");
  }
  __syncthreads();

  // Build A fragments from LDS (ds_load_b128), shared by all 16 waves.
  FragBF a[8];
  #pragma unroll
  for (int ks = 0; ks < 8; ++ks) a[ks] = load_a_frag(aRows, lr, ks * 32, lane);

  float bv[5]; int bi[5];
  #pragma unroll
  for (int t = 0; t < 5; ++t) { bv[t] = -INFINITY; bi[t] = 0x7FFFFFFF; }

  for (int pass = 0; pass < 8; ++pass) {
    int pbase = pass * 512;
    // GEMM: this wave computes 2 column tiles of the 16x512 strip.
    for (int t = 0; t < 2; ++t) {
      int m0 = pbase + (wave * 2 + t) * 16;
      // Batch all 8 B fragments into distinct registers (one load clause),
      // then issue 8 back-to-back WMMAs. sched_barrier(0) pins the split so
      // the scheduler cannot sink loads next to their WMMA consumers.
      FragBF bf[8];
      #pragma unroll
      for (int ks = 0; ks < 8; ++ks) bf[ks] = load_b_frag(xtb, m0 + lr, ks * 32, lane);
      __builtin_amdgcn_sched_barrier(0);
      v8f acc = {};
      #pragma unroll
      for (int ks = 0; ks < 8; ++ks)
        acc = __builtin_amdgcn_wmma_f32_16x16x32_bf16(
            false, a[ks].v, false, bf[ks].v, (short)0, acc, false, false);
      float xxv = xx[b * N_ + m0 + lr];  // -xx[row] is constant per row: drop
      #pragma unroll
      for (int i = 0; i < 8; ++i) {
        int row = i + half * 8;
        strip[row * 512 + (m0 - pbase) + lr] = 2.0f * acc[i] - xxv;
      }
    }
    __syncthreads();
    // scan: wave r owns row r; maintain running top-5 (jax tie: lower index)
    {
      int r = wave;
      #pragma unroll
      for (int j = 0; j < 16; ++j) {
        int lc = lane + 32 * j;
        float v = strip[r * 512 + lc];
        int gi = pbase + lc;
        if ((v > bv[4]) || (v == bv[4] && gi < bi[4])) {
          bv[4] = v; bi[4] = gi;
          #pragma unroll
          for (int t = 4; t >= 1; --t) {
            bool sw = (bv[t] > bv[t - 1]) || (bv[t] == bv[t - 1] && bi[t] < bi[t - 1]);
            if (sw) {
              float tv = bv[t]; bv[t] = bv[t - 1]; bv[t - 1] = tv;
              int   ti = bi[t]; bi[t] = bi[t - 1]; bi[t - 1] = ti;
            }
          }
        }
      }
    }
    __syncthreads();
  }
  #pragma unroll
  for (int t = 0; t < 5; ++t) { candV[wave][lane][t] = bv[t]; candI[wave][lane][t] = bi[t]; }
  __syncthreads();
  if (lane == 0) {
    int outi[5];
    for (int sel = 0; sel < 5; ++sel) {
      float best = -INFINITY; int besti = 0x7FFFFFFF; int bl = 0, bt = 0;
      for (int l = 0; l < 32; ++l)
        for (int t = 0; t < 5; ++t) {
          float v = candV[wave][l][t]; int gi = candI[wave][l][t];
          if ((v > best) || (v == best && gi < besti)) { best = v; besti = gi; bl = l; bt = t; }
        }
      outi[sel] = besti;
      candV[wave][bl][bt] = -INFINITY; candI[wave][bl][bt] = 0x7FFFFFFF;
    }
    int n = n0 + wave;
    for (int t = 0; t < 5; ++t) idxo[((size_t)b * N_ + n) * K_ + t] = outi[t];
  }
}

// ---- P = V*X, Q = U*X: preload all fragments, 16 back-to-back WMMAs ----
__global__ void __launch_bounds__(128) k_feat_gemm(
    const unsigned short* __restrict__ xt, const unsigned short* __restrict__ Vb,
    const unsigned short* __restrict__ Ub, float* __restrict__ P, float* __restrict__ Q) {
  int bx = blockIdx.x;
  int b   = bx / (16 * 64);
  int rem = bx % (16 * 64);
  int o0  = (rem / 64) * 16;
  int n0  = (rem % 64) * 64;
  int wave = threadIdx.x >> 5, lane = threadIdx.x & 31;
  int lr = lane & 15, half = lane >> 4;
  int nw = n0 + wave * 16;
  const unsigned short* xtb = xt + (size_t)b * N_ * C_;

  FragBF av[8], au[8], bf[8];
  #pragma unroll
  for (int ks = 0; ks < 8; ++ks) {
    int kb = ks * 32;
    av[ks] = load_a_frag(Vb, o0 + lr, kb, lane);
    au[ks] = load_a_frag(Ub, o0 + lr, kb, lane);
    bf[ks] = load_b_frag(xtb, nw + lr, kb, lane);
  }
  __builtin_amdgcn_sched_barrier(0);
  v8f accP = {}, accQ = {};
  #pragma unroll
  for (int ks = 0; ks < 8; ++ks) {
    accP = __builtin_amdgcn_wmma_f32_16x16x32_bf16(false, av[ks].v, false, bf[ks].v, (short)0, accP, false, false);
    accQ = __builtin_amdgcn_wmma_f32_16x16x32_bf16(false, au[ks].v, false, bf[ks].v, (short)0, accQ, false, false);
  }
  #pragma unroll
  for (int i = 0; i < 8; ++i) {
    int o = o0 + i + half * 8;
    int n = nw + lr;
    P[((size_t)b * O_ + o) * N_ + n] = accP[i];
    Q[((size_t)b * O_ + o) * N_ + n] = accQ[i];
  }
}

// ---- BN stats: per-channel sum / sumsq of y = P[gather] + Q ----
__global__ void __launch_bounds__(256) k_stats(
    const float* __restrict__ P, const float* __restrict__ Q,
    const int* __restrict__ idx, float* __restrict__ sums, float* __restrict__ sumsq) {
  int o = blockIdx.x >> 5;
  int chunk = blockIdx.x & 31;
  int tid = chunk * 256 + threadIdx.x;  // 0..8191
  float s = 0.f, s2 = 0.f;
  for (int it = 0; it < 2; ++it) {
    int g = tid + it * 8192;            // b*N + n
    int b = g >> 12, n = g & (N_ - 1);
    float q = Q[((size_t)b * O_ + o) * N_ + n];
    const int* ip = idx + (size_t)(b * N_ + n) * K_;
    #pragma unroll
    for (int k = 0; k < K_; ++k) {
      float y = P[((size_t)b * O_ + o) * N_ + ip[k]] + q;
      s += y; s2 += y * y;
    }
  }
  __shared__ float rs[256], rs2[256];
  rs[threadIdx.x] = s; rs2[threadIdx.x] = s2;
  __syncthreads();
  for (int off = 128; off > 0; off >>= 1) {
    if (threadIdx.x < off) {
      rs[threadIdx.x]  += rs[threadIdx.x + off];
      rs2[threadIdx.x] += rs2[threadIdx.x + off];
    }
    __syncthreads();
  }
  if (threadIdx.x == 0) { atomicAdd(&sums[o], rs[0]); atomicAdd(&sumsq[o], rs2[0]); }
}

// ---- fold BN into y*a + b ----
__global__ void k_scale(const float* __restrict__ sums, const float* __restrict__ sumsq,
                        const float* __restrict__ gamma, const float* __restrict__ beta,
                        float* __restrict__ an, float* __restrict__ bn) {
  int o = threadIdx.x;
  const float cnt = (float)(B_ * N_ * K_);
  float m   = sums[o] / cnt;
  float var = sumsq[o] / cnt - m * m;
  float g   = gamma[o] * rsqrtf(var + BN_EPS);
  an[o] = g;
  bn[o] = beta[o] - m * g;
}

// ---- epilogue: relu(max_k(a*y+b)) + x ----
__global__ void __launch_bounds__(256) k_final(
    const float* __restrict__ P, const float* __restrict__ Q,
    const int* __restrict__ idx, const float* __restrict__ an,
    const float* __restrict__ bn, const float* __restrict__ x,
    float* __restrict__ out) {
  int i = blockIdx.x * 256 + threadIdx.x;   // (b*O + o)*N + n
  int n  = i & (N_ - 1);
  int bo = i >> 12;
  int o  = bo & (O_ - 1);
  int b  = bo >> 8;
  float av = an[o], bvv = bn[o];
  float q = Q[i];
  const int* ip = idx + (size_t)(b * N_ + n) * K_;
  float m = -INFINITY;
  #pragma unroll
  for (int k = 0; k < K_; ++k)
    m = fmaxf(m, av * (P[(size_t)bo * N_ + ip[k]] + q) + bvv);
  out[i] = fmaxf(m, 0.f) + x[i];
}

extern "C" void kernel_launch(void* const* d_in, const int* in_sizes, int n_in,
                              void* d_out, int out_size, void* d_ws, size_t ws_size,
                              hipStream_t stream) {
  const float* x     = (const float*)d_in[0];
  const float* W     = (const float*)d_in[1];
  const float* gamma = (const float*)d_in[2];
  const float* beta  = (const float*)d_in[3];
  float* out = (float*)d_out;

  char* ws = (char*)d_ws;
  size_t off = 0;
  auto alloc = [&](size_t bytes) {
    void* p = ws + off;
    off += (bytes + 255) & ~(size_t)255;
    return p;
  };
  unsigned short* xt = (unsigned short*)alloc((size_t)B_ * N_ * C_ * 2);
  unsigned short* Vb = (unsigned short*)alloc((size_t)O_ * C_ * 2);
  unsigned short* Ub = (unsigned short*)alloc((size_t)O_ * C_ * 2);
  float* xx    = (float*)alloc((size_t)B_ * N_ * 4);
  int*   idx   = (int*)alloc((size_t)B_ * N_ * K_ * 4);
  float* P     = (float*)alloc((size_t)B_ * O_ * N_ * 4);
  float* Q     = (float*)alloc((size_t)B_ * O_ * N_ * 4);
  float* sums  = (float*)alloc(256 * 4);   // contiguous with sumsq (256B aligned)
  float* sumsq = (float*)alloc(256 * 4);
  float* an    = (float*)alloc(256 * 4);
  float* bn    = (float*)alloc(256 * 4);

  k_xt<<<dim3(N_ / 32, C_ / 32, B_), dim3(32, 8), 0, stream>>>(x, xt);
  k_xx<<<B_ * N_ / 256, 256, 0, stream>>>(x, xx);
  k_w<<<O_, C_, 0, stream>>>(W, Vb, Ub);
  k_zero<<<2, 256, 0, stream>>>(sums, 512);  // zeros sums + sumsq

  k_dist_topk<<<B_ * (N_ / 16), 512, 0, stream>>>(xt, xx, idx);
  k_feat_gemm<<<B_ * 16 * 64, 128, 0, stream>>>(xt, Vb, Ub, P, Q);
  k_stats<<<256 * 32, 256, 0, stream>>>(P, Q, idx, sums, sumsq);
  k_scale<<<1, 256, 0, stream>>>(sums, sumsq, gamma, beta, an, bn);
  k_final<<<B_ * O_ * N_ / 256, 256, 0, stream>>>(P, Q, idx, an, bn, x, out);
}
